// GeoRegionSampler_3487513444477
// MI455X (gfx1250) — matrix operand order repaired and checked by reading, not compiled, not run on hardware
//
#include <hip/hip_runtime.h>
#include <hip/hip_bf16.h>

// ---------------------------------------------------------------------------
// CDNA5 WMMA (wave32): D(16x16 f32) = A(16x32 bf16) x B(32x16 bf16) + C
// ---------------------------------------------------------------------------
typedef __attribute__((ext_vector_type(16))) __bf16 v16bf;
typedef __attribute__((ext_vector_type(8)))  float  v8f;

#define CCH   1024   // feature channels C
#define DP    1026   // C + 2
#define DPAD  1056   // DP padded to multiple of 32 (33 K-steps, 66 N-tiles)
#define KNEI  24     // neighbors per center
#define RROWS 32     // KNEI padded to 2 M-tiles

// ---------------------------------------------------------------------------
// Branch-free fragment loaders on pre-padded bf16 data (ISA 7.12.2 layouts).
// A 16x32: lane<16 -> kb=0, lane>=16 -> kb=8; elems 0-7 = r[kb..kb+7],
//          elems 8-15 = r[kb+16..kb+23]  -> two 16B vector loads.
// B 32x16: B[k][n] = W[n][k]; lane col = lane&15, kb = (lane<16)?0:8;
//          elems 0-7 = row n [kb..kb+7], elems 8-15 = [kb+16..kb+23] -> two 16B vector loads.
// ---------------------------------------------------------------------------
__device__ __forceinline__ v16bf load_a_frag_bf(const __bf16* base, int lda,
                                                int m0, int k0) {
  int lane = threadIdx.x & 31;
  const __bf16* r = base + (size_t)(m0 + (lane & 15)) * lda + k0 +
                    ((lane < 16) ? 0 : 8);
  union { uint4 u[2]; v16bf v; } t;
  t.u[0] = *(const uint4*)(r);
  t.u[1] = *(const uint4*)(r + 16);
  return t.v;
}

__device__ __forceinline__ v16bf load_b_frag_bf(const __bf16* W, int ldw,
                                                int n0, int k0) {
  int lane = threadIdx.x & 31;
  const __bf16* r = W + (size_t)(n0 + (lane & 15)) * ldw + k0 +
                    ((lane < 16) ? 0 : 8);
  union { uint4 u[2]; v16bf v; } t;
  t.u[0] = *(const uint4*)(r);
  t.u[1] = *(const uint4*)(r + 16);
  return t.v;
}

// ---------------------------------------------------------------------------
// 0) Grid-stride convert + zero-pad: f32 (nrows x ncols, stride) -> bf16
//    (drows x dcols).
// ---------------------------------------------------------------------------
__global__ void cvt_pad_kernel(const float* __restrict__ src, __bf16* __restrict__ dst,
                               int nrows, int ncols, int sstride,
                               int drows, int dcols) {
  size_t total = (size_t)drows * dcols;
  for (size_t i = (size_t)blockIdx.x * blockDim.x + threadIdx.x; i < total;
       i += (size_t)gridDim.x * blockDim.x) {
    int r = (int)(i / dcols);
    int c = (int)(i - (size_t)r * dcols);
    float v = (r < nrows && c < ncols) ? src[(size_t)r * sstride + c] : 0.0f;
    dst[i] = (__bf16)v;
  }
}

// ---------------------------------------------------------------------------
// 1) Bilinear sampling: all_fea[b,n,c]
// ---------------------------------------------------------------------------
__global__ void bilinear_kernel(const float* __restrict__ fmap,
                                const float* __restrict__ pts,
                                float* __restrict__ fea) {
  const int H = 24, W = 24;
  int bn = blockIdx.x;                 // b*512 + n
  float py = pts[bn * 2 + 0];
  float px = pts[bn * 2 + 1];
  float gy = py * (H - 1), gx = px * (W - 1);
  float fy = floorf(gy), fx = floorf(gx);
  float wy = gy - fy, wx = gx - fx;
  int y0 = min(max((int)fy, 0), H - 1);
  int y1 = min(y0 + 1, H - 1);
  int x0 = min(max((int)fx, 0), W - 1);
  int x1 = min(x0 + 1, W - 1);
  float w00 = (1.f - wy) * (1.f - wx), w01 = (1.f - wy) * wx;
  float w10 = wy * (1.f - wx),         w11 = wy * wx;
  int b = bn >> 9;                     // N0 = 512
  const float* fb = fmap + (size_t)b * CCH * H * W;
  for (int c = threadIdx.x; c < CCH; c += 256) {
    const float* fc = fb + (size_t)c * H * W;
    float v = fc[y0 * W + x0] * w00 + fc[y0 * W + x1] * w01 +
              fc[y1 * W + x0] * w10 + fc[y1 * W + x1] * w11;
    fea[(size_t)bn * CCH + c] = v;
  }
}

// ---------------------------------------------------------------------------
// 2) Farthest point sampling: one block (512 thr) per batch.
// ---------------------------------------------------------------------------
__global__ void fps_kernel(const float* __restrict__ pts, int N, int S,
                           int* __restrict__ fi) {
  int b = blockIdx.x;
  __shared__ float px[512], py[512];
  __shared__ float rv[16];
  __shared__ int   ri[16];
  __shared__ int   far_s;
  int t = threadIdx.x;
  bool act = t < N;
  float x = 0.f, y = 0.f, mind = 1e10f;
  if (act) {
    x = pts[((size_t)b * N + t) * 2 + 0];
    y = pts[((size_t)b * N + t) * 2 + 1];
    px[t] = x; py[t] = y;
  }
  if (t == 0) far_s = 0;
  __syncthreads();
  for (int i = 0; i < S; ++i) {
    int far = far_s;
    if (t == 0) fi[b * S + i] = far;
    float cx = px[far], cy = py[far];
    if (act) {
      float dx = x - cx, dy = y - cy;
      mind = fminf(mind, dx * dx + dy * dy);
    }
    float v = act ? mind : -1.0f;
    int   id = t;
#pragma unroll
    for (int off = 16; off > 0; off >>= 1) {
      float ov = __shfl_down(v, off, 32);
      int   oi = __shfl_down(id, off, 32);
      if (ov > v || (ov == v && oi < id)) { v = ov; id = oi; }
    }
    int w = t >> 5;
    if ((t & 31) == 0) { rv[w] = v; ri[w] = id; }
    __syncthreads();
    if (t == 0) {
      float bv = rv[0]; int bi = ri[0];
      for (int q = 1; q < 16; ++q)
        if (rv[q] > bv || (rv[q] == bv && ri[q] < bi)) { bv = rv[q]; bi = ri[q]; }
      far_s = bi;
    }
    __syncthreads();
  }
}

// ---------------------------------------------------------------------------
// 3) kNN: one block (128 thr) per (b,s); K smallest; writes new_pts.
// ---------------------------------------------------------------------------
__global__ void knn_kernel(const float* __restrict__ pts, const int* __restrict__ fi,
                           int N, int S, int* __restrict__ nidx,
                           float* __restrict__ new_pts) {
  int bs = blockIdx.x;
  int b  = bs / S;
  __shared__ float d2[512];
  __shared__ float rv[4];
  __shared__ int   ri[4];
  int t = threadIdx.x;
  int anc = fi[bs];
  float qx = pts[((size_t)b * N + anc) * 2 + 0];
  float qy = pts[((size_t)b * N + anc) * 2 + 1];
  if (t == 0) { new_pts[bs * 2 + 0] = qx; new_pts[bs * 2 + 1] = qy; }
  for (int n = t; n < N; n += 128) {
    float dx = pts[((size_t)b * N + n) * 2 + 0] - qx;
    float dy = pts[((size_t)b * N + n) * 2 + 1] - qy;
    d2[n] = dx * dx + dy * dy;
  }
  __syncthreads();
  for (int k = 0; k < KNEI; ++k) {
    float v = 1e30f; int id = 0;
    for (int n = t; n < N; n += 128)
      if (d2[n] < v) { v = d2[n]; id = n; }
#pragma unroll
    for (int off = 16; off > 0; off >>= 1) {
      float ov = __shfl_down(v, off, 32);
      int   oi = __shfl_down(id, off, 32);
      if (ov < v || (ov == v && oi < id)) { v = ov; id = oi; }
    }
    if ((t & 31) == 0) { rv[t >> 5] = v; ri[t >> 5] = id; }
    __syncthreads();
    if (t == 0) {
      float bv = rv[0]; int bi = ri[0];
      for (int q = 1; q < 4; ++q)
        if (rv[q] < bv || (rv[q] == bv && ri[q] < bi)) { bv = rv[q]; bi = ri[q]; }
      nidx[bs * KNEI + k] = bi;
      d2[bi] = 1e30f;
    }
    __syncthreads();
  }
}

// ---------------------------------------------------------------------------
// 4) Fused region MLP: one 256-thread workgroup per (b,s).
//    Weights arrive pre-padded bf16 -> the GEMM K-loops are branch-free:
//    4x ds_load_b128 (A) + 2x global_load_b128 (B) + 2x v_wmma per step.
// ---------------------------------------------------------------------------
__global__ __launch_bounds__(256)
void region_kernel(const float* __restrict__ fea_in, const float* __restrict__ pts_in,
                   const int* __restrict__ fi, const int* __restrict__ nidx,
                   const __bf16* __restrict__ Wd_bf,   // DPAD x DPAD
                   const float* __restrict__ bd,
                   const __bf16* __restrict__ Wa1_bf,  // CCH x DPAD (first half)
                   const float* __restrict__ Wa,       // f32, for anchor proj
                   const float* __restrict__ ba,
                   const float* __restrict__ g, const float* __restrict__ beta,
                   float* __restrict__ fea_out, int Nin, int S) {
  extern __shared__ char smem_raw[];
  __bf16* Abuf  = (__bf16*)smem_raw;            // RROWS*DPAD
  __bf16* Dbuf  = Abuf + RROWS * DPAD;          // RROWS*DPAD
  float*  aproj = (float*)(Dbuf + RROWS * DPAD);
  float*  anch  = aproj + CCH;                  // DP (+2 pad)
  float*  red   = anch + DP + 2;                // 16
  __bf16* hbuf  = Abuf;                         // reuse A region: KNEI x CCH

  int bs = blockIdx.x;
  int b  = bs / S;
  int t  = threadIdx.x;
  int w  = t >> 5, lane = t & 31;
  int aid = fi[bs];

  // stage anchor vector (f32 in LDS)
  for (int d = t; d < DP; d += 256)
    anch[d] = (d < CCH) ? fea_in[((size_t)b * Nin + aid) * CCH + d]
                        : pts_in[((size_t)b * Nin + aid) * 2 + (d - CCH)];
  // zero-fill Abuf (covers all padding), 8 bf16 per 16B store
  for (int i = t; i < (RROWS * DPAD) / 8; i += 256)
    ((uint4*)Abuf)[i] = make_uint4(0u, 0u, 0u, 0u);
  __syncthreads();

  // stage A rows: features (4-wide chunks, packed 8B LDS stores)
  for (int i = t; i < KNEI * (CCH / 4); i += 256) {
    int r = i / (CCH / 4);
    int d = (i - r * (CCH / 4)) * 4;
    int nid = nidx[bs * KNEI + r];
    const float* src = fea_in + ((size_t)b * Nin + nid) * CCH + d;
    float4 f = *(const float4*)src;
    __bf16 tmp[4];
    tmp[0] = (__bf16)(f.x - anch[d + 0]);
    tmp[1] = (__bf16)(f.y - anch[d + 1]);
    tmp[2] = (__bf16)(f.z - anch[d + 2]);
    tmp[3] = (__bf16)(f.w - anch[d + 3]);
    *(uint2*)(Abuf + r * DPAD + d) = *(uint2*)tmp;
  }
  // stage A rows: the two point coords
  if (t < KNEI * 2) {
    int r = t >> 1, d = CCH + (t & 1);
    int nid = nidx[bs * KNEI + r];
    float lv = pts_in[((size_t)b * Nin + nid) * 2 + (t & 1)];
    Abuf[r * DPAD + d] = (__bf16)(lv - anch[d]);
  }
  // anchor projection through second half of W_agg (shared across all k rows)
  for (int c = t; c < CCH; c += 256) {
    const float* wr = Wa + (size_t)c * (2 * DP) + DP;
    float acc = 0.f;
    for (int d = 0; d < DP; ++d) acc += anch[d] * wr[d];
    aproj[c] = acc;
  }
  __syncthreads();

  // ---- GEMM1: Dbuf = Abuf @ Wd^T (+bd on valid cols); 2 M-tiles share B ----
  for (int nt = w; nt < DPAD / 16; nt += 8) {
    v8f acc0 = {}, acc1 = {};
    for (int ks = 0; ks < DPAD; ks += 32) {
      v16bf bf = load_b_frag_bf(Wd_bf, DPAD, nt * 16, ks);
      v16bf a0 = load_a_frag_bf(Abuf, DPAD, 0,  ks);
      v16bf a1 = load_a_frag_bf(Abuf, DPAD, 16, ks);
      acc0 = __builtin_amdgcn_wmma_f32_16x16x32_bf16(false, a0, false, bf,
                                                     (short)0, acc0, false, false);
      acc1 = __builtin_amdgcn_wmma_f32_16x16x32_bf16(false, a1, false, bf,
                                                     (short)0, acc1, false, false);
    }
    int col = nt * 16 + (lane & 15);
    float bias = (col < DP) ? bd[col] : 0.f;   // pad cols: acc==0, bias==0
#pragma unroll
    for (int v = 0; v < 8; ++v) {
      int row = (lane < 16) ? v : v + 8;
      Dbuf[row * DPAD + col]        = (__bf16)(acc0[v] + bias);
      Dbuf[(row + 16) * DPAD + col] = (__bf16)(acc1[v] + bias);
    }
  }
  __syncthreads();

  // ---- GEMM2: h = relu(Dbuf @ Wagg1^T + aproj + ba) -> hbuf (bf16) ----
  for (int nt = w; nt < CCH / 16; nt += 8) {
    v8f acc0 = {}, acc1 = {};
    for (int ks = 0; ks < DPAD; ks += 32) {
      v16bf bf = load_b_frag_bf(Wa1_bf, DPAD, nt * 16, ks);
      v16bf a0 = load_a_frag_bf(Dbuf, DPAD, 0,  ks);
      v16bf a1 = load_a_frag_bf(Dbuf, DPAD, 16, ks);
      acc0 = __builtin_amdgcn_wmma_f32_16x16x32_bf16(false, a0, false, bf,
                                                     (short)0, acc0, false, false);
      acc1 = __builtin_amdgcn_wmma_f32_16x16x32_bf16(false, a1, false, bf,
                                                     (short)0, acc1, false, false);
    }
    int col = nt * 16 + (lane & 15);
    float add = aproj[col] + ba[col];
#pragma unroll
    for (int v = 0; v < 8; ++v) {
      int row = (lane < 16) ? v : v + 8;
      hbuf[row * CCH + col] = (__bf16)fmaxf(acc0[v] + add, 0.f);
      int row1 = row + 16;
      if (row1 < KNEI)
        hbuf[row1 * CCH + col] = (__bf16)fmaxf(acc1[v] + add, 0.f);
    }
  }
  __syncthreads();

  // ---- LayerNorm per row + mean over rows ----
  float accout[4] = {0.f, 0.f, 0.f, 0.f};
  for (int m = 0; m < KNEI; ++m) {
    float s1 = 0.f, s2 = 0.f;
#pragma unroll
    for (int j = 0; j < 4; ++j) {
      float hv = (float)hbuf[m * CCH + t + j * 256];
      s1 += hv; s2 += hv * hv;
    }
#pragma unroll
    for (int off = 16; off > 0; off >>= 1) {
      s1 += __shfl_down(s1, off, 32);
      s2 += __shfl_down(s2, off, 32);
    }
    if (lane == 0) { red[w] = s1; red[8 + w] = s2; }
    __syncthreads();
    float ts1 = 0.f, ts2 = 0.f;
#pragma unroll
    for (int q = 0; q < 8; ++q) { ts1 += red[q]; ts2 += red[8 + q]; }
    float mean = ts1 * (1.f / CCH);
    float var  = ts2 * (1.f / CCH) - mean * mean;
    float inv  = rsqrtf(var + 1e-5f);
#pragma unroll
    for (int j = 0; j < 4; ++j) {
      float hv = (float)hbuf[m * CCH + t + j * 256];
      accout[j] += (hv - mean) * inv;
    }
    __syncthreads();
  }
#pragma unroll
  for (int j = 0; j < 4; ++j) {
    int c = t + j * 256;
    fea_out[(size_t)bs * CCH + c] = accout[j] * (1.f / KNEI) * g[c] + beta[c];
  }
}

// ---------------------------------------------------------------------------
// 5) Tail GEMM: D(32 x N) = A(32 x K bf16) @ B(N x K bf16)^T + bias.
//    One wave per N-tile; both M-tiles share each B fragment.
// ---------------------------------------------------------------------------
__global__ __launch_bounds__(32)
void gemm_nt_bf_kernel(const __bf16* __restrict__ A, const __bf16* __restrict__ Bw,
                       const float* __restrict__ bias, float* __restrict__ D,
                       int N, int K) {
  int nt = blockIdx.x;
  v8f acc0 = {}, acc1 = {};
  for (int k0 = 0; k0 < K; k0 += 32) {
    v16bf bf = load_b_frag_bf(Bw, K, nt * 16, k0);
    v16bf a0 = load_a_frag_bf(A, K, 0,  k0);
    v16bf a1 = load_a_frag_bf(A, K, 16, k0);
    acc0 = __builtin_amdgcn_wmma_f32_16x16x32_bf16(false, a0, false, bf,
                                                   (short)0, acc0, false, false);
    acc1 = __builtin_amdgcn_wmma_f32_16x16x32_bf16(false, a1, false, bf,
                                                   (short)0, acc1, false, false);
  }
  int lane = threadIdx.x & 31;
  int col  = nt * 16 + (lane & 15);
  float bv = bias[col];
#pragma unroll
  for (int v = 0; v < 8; ++v) {
    int row = (lane < 16) ? v : v + 8;
    D[(size_t)row * N + col]        = acc0[v] + bv;
    D[(size_t)(row + 16) * N + col] = acc1[v] + bv;
  }
}

// ---------------------------------------------------------------------------
// Host launcher
// ---------------------------------------------------------------------------
extern "C" void kernel_launch(void* const* d_in, const int* in_sizes, int n_in,
                              void* d_out, int out_size, void* d_ws, size_t ws_size,
                              hipStream_t stream) {
  const float* fmap   = (const float*)d_in[0];
  const float* points = (const float*)d_in[1];
  const float* W_diff = (const float*)d_in[2];
  const float* b_diff = (const float*)d_in[3];
  const float* W_agg  = (const float*)d_in[4];
  const float* b_agg  = (const float*)d_in[5];
  const float* ln_g   = (const float*)d_in[6];
  const float* ln_b   = (const float*)d_in[7];
  const float* W_flat = (const float*)d_in[8];
  const float* b_flat = (const float*)d_in[9];
  const float* W_dim  = (const float*)d_in[10];
  const float* b_dim  = (const float*)d_in[11];
  float* out = (float*)d_out;
  (void)in_sizes; (void)n_in; (void)out_size; (void)ws_size;

  const int B = 32, N0 = 512;
  float* ws = (float*)d_ws;
  size_t off = 0;
  float* fea0 = ws + off; off += (size_t)B * N0 * CCH;
  float* fea1 = ws + off; off += (size_t)B * 128 * CCH;
  float* fea2 = ws + off; off += (size_t)B * 32 * CCH;
  float* pts1 = ws + off; off += (size_t)B * 128 * 2;
  float* pts2 = ws + off; off += (size_t)B * 32 * 2;
  float* xbuf = ws + off; off += (size_t)B * CCH;
  int* fi   = (int*)(ws + off); off += (size_t)B * 128;
  int* nidx = (int*)(ws + off); off += (size_t)B * 128 * KNEI;

  __bf16* bws = (__bf16*)(ws + off);
  size_t boff = 0;
  __bf16* Wd_bf   = bws + boff; boff += (size_t)2 * DPAD * DPAD;
  __bf16* Wa1_bf  = bws + boff; boff += (size_t)2 * CCH * DPAD;
  __bf16* Wflat_bf= bws + boff; boff += (size_t)CCH * 32 * CCH;
  __bf16* Wdim_bf = bws + boff; boff += (size_t)4096 * CCH;
  __bf16* fea2_bf = bws + boff; boff += (size_t)B * 32 * CCH;
  __bf16* xbuf_bf = bws + boff; boff += (size_t)B * CCH;

  size_t lds_bytes = (size_t)2 * RROWS * DPAD * sizeof(__bf16) +
                     (size_t)(CCH + DP + 2 + 16) * sizeof(float);

  // ---- weight pre-conversion to padded bf16 (branch-free GEMM K-loops) ----
  cvt_pad_kernel<<<512, 256, 0, stream>>>(W_diff, Wd_bf, DP, DP, DP, DPAD, DPAD);
  cvt_pad_kernel<<<512, 256, 0, stream>>>(W_diff + (size_t)DP * DP,
                                          Wd_bf + (size_t)DPAD * DPAD,
                                          DP, DP, DP, DPAD, DPAD);
  cvt_pad_kernel<<<512, 256, 0, stream>>>(W_agg, Wa1_bf, CCH, DP, 2 * DP, CCH, DPAD);
  cvt_pad_kernel<<<512, 256, 0, stream>>>(W_agg + (size_t)CCH * 2 * DP,
                                          Wa1_bf + (size_t)CCH * DPAD,
                                          CCH, DP, 2 * DP, CCH, DPAD);
  cvt_pad_kernel<<<4096, 256, 0, stream>>>(W_flat, Wflat_bf, CCH, 32 * CCH,
                                           32 * CCH, CCH, 32 * CCH);
  cvt_pad_kernel<<<1024, 256, 0, stream>>>(W_dim, Wdim_bf, 4096, CCH, CCH, 4096, CCH);

  // ---- stage 0: bilinear features ----
  bilinear_kernel<<<B * N0, 256, 0, stream>>>(fmap, points, fea0);

  // ---- layer 0: 512 -> 128 centers ----
  fps_kernel<<<B, 512, 0, stream>>>(points, 512, 128, fi);
  knn_kernel<<<B * 128, 128, 0, stream>>>(points, fi, 512, 128, nidx, pts1);
  region_kernel<<<B * 128, 256, lds_bytes, stream>>>(
      fea0, points, fi, nidx,
      Wd_bf, b_diff, Wa1_bf, W_agg, b_agg, ln_g, ln_b,
      fea1, 512, 128);

  // ---- layer 1: 128 -> 32 centers ----
  fps_kernel<<<B, 512, 0, stream>>>(pts1, 128, 32, fi);
  knn_kernel<<<B * 32, 128, 0, stream>>>(pts1, fi, 128, 32, nidx, pts2);
  region_kernel<<<B * 32, 256, lds_bytes, stream>>>(
      fea1, pts1, fi, nidx,
      Wd_bf + (size_t)DPAD * DPAD, b_diff + DP,
      Wa1_bf + (size_t)CCH * DPAD, W_agg + (size_t)CCH * 2 * DP, b_agg + CCH,
      ln_g + CCH, ln_b + CCH,
      fea2, 128, 32);

  // ---- tail: x = fea2.reshape(32,32768) @ W_flat^T + b_flat; out = x @ W_dim^T + b_dim
  cvt_pad_kernel<<<512, 256, 0, stream>>>(fea2, fea2_bf, 32, 32 * CCH, 32 * CCH,
                                          32, 32 * CCH);
  gemm_nt_bf_kernel<<<CCH / 16, 32, 0, stream>>>(fea2_bf, Wflat_bf, b_flat, xbuf,
                                                 CCH, 32 * CCH);
  cvt_pad_kernel<<<64, 256, 0, stream>>>(xbuf, xbuf_bf, 32, CCH, CCH, 32, CCH);
  gemm_nt_bf_kernel<<<4096 / 16, 32, 0, stream>>>(xbuf_bf, Wdim_bf, b_dim, out,
                                                  4096, CCH);
}
